// GAT_Byzantine_Detector_27874337751516
// MI455X (gfx1250) — matrix operand order repaired and checked by reading, not compiled, not run on hardware
//
#include <hip/hip_runtime.h>
#include <hip/hip_bf16.h>

// ---------------- problem constants ----------------
#define BB 4
#define NN 10000
#define TT 128
#define EE 160000
#define HH 4
#define CC 64
#define GD 256          // H*C
#define BN (BB*NN)      // 40000
#define EALL (EE+NN)    // edges + self loops per graph

typedef __bf16 bf16_t;
typedef __attribute__((ext_vector_type(8)))  __bf16 v8bf;
typedef __attribute__((ext_vector_type(16))) __bf16 v16bf;
typedef __attribute__((ext_vector_type(8)))  float  v8f;

// ---------------- helpers ----------------
__device__ __forceinline__ unsigned short f2bf(float f) {
    unsigned u = __float_as_uint(f);
    unsigned r = (u + 0x7FFFu + ((u >> 16) & 1u)) >> 16;
    return (unsigned short)r;
}
__device__ __forceinline__ unsigned f2ord(float f) {
    unsigned u = __float_as_uint(f);
    return (u & 0x80000000u) ? ~u : (u | 0x80000000u);
}
__device__ __forceinline__ float ord2f(unsigned u) {
    unsigned v = (u & 0x80000000u) ? (u & 0x7FFFFFFFu) : ~u;
    return __uint_as_float(v);
}
__device__ __forceinline__ void atomAddF(float* p, float v) {
    unsafeAtomicAdd(p, v);   // native global_atomic_add_f32
}
__device__ __forceinline__ float lrelu(float v) { return v > 0.f ? v : 0.2f * v; }

// ---------------- transpose fp32 (K,N) -> bf16 (N,K) ----------------
__global__ void k_transpose_bf16(const float* __restrict__ in, unsigned short* __restrict__ out,
                                 int K, int N) {
    int idx = blockIdx.x * blockDim.x + threadIdx.x;
    if (idx >= K * N) return;
    int k = idx / N, n = idx % N;
    out[n * K + k] = f2bf(in[idx]);
}

// ---------------- temporal conv: x(BN,128) -> featbf(BN,64) ----------------
__global__ void __launch_bounds__(128)
k_conv_temporal(const float* __restrict__ x,
                const float* __restrict__ w1, const float* __restrict__ b1,
                const float* __restrict__ w2, const float* __restrict__ b2,
                unsigned short* __restrict__ featbf) {
    __shared__ float sx[TT];
    __shared__ float w2s[64 * 32 * 3];
    __shared__ float h1[32 * TT];
    __shared__ float pl[32 * 64];
    __shared__ float part[128];
    const int tid = threadIdx.x;
    const int node = blockIdx.x;

    sx[tid] = x[(size_t)node * TT + tid];
    for (int i = tid; i < 64 * 32 * 3; i += 128) w2s[i] = w2[i];
    __syncthreads();

    // conv1 (1->32, k=3, pad=1) + relu
    const int pos = tid;
    float xm1 = (pos > 0)       ? sx[pos - 1] : 0.f;
    float xc  = sx[pos];
    float xp1 = (pos < TT - 1)  ? sx[pos + 1] : 0.f;
    for (int ch = 0; ch < 32; ++ch) {
        float v = w1[ch * 3 + 0] * xm1 + w1[ch * 3 + 1] * xc + w1[ch * 3 + 2] * xp1 + b1[ch];
        h1[ch * TT + pos] = fmaxf(v, 0.f);
    }
    __syncthreads();

    // maxpool k=2 s=2 -> (32,64)
    for (int i = tid; i < 32 * 64; i += 128) {
        int ch = i >> 6, j = i & 63;
        pl[i] = fmaxf(h1[ch * TT + 2 * j], h1[ch * TT + 2 * j + 1]);
    }
    __syncthreads();

    // conv2 (32->64, k=3, pad=1) + relu + mean over 64 positions
    const int cout = tid >> 1;
    const int jb   = (tid & 1) * 32;
    float acc = 0.f;
    for (int jj = 0; jj < 32; ++jj) {
        int j = jb + jj;
        float s = b2[cout];
        const float* wrow = &w2s[cout * 96];
        for (int ci = 0; ci < 32; ++ci) {
            float pm = (j > 0)  ? pl[ci * 64 + j - 1] : 0.f;
            float pc = pl[ci * 64 + j];
            float pp = (j < 63) ? pl[ci * 64 + j + 1] : 0.f;
            s += wrow[ci * 3 + 0] * pm + wrow[ci * 3 + 1] * pc + wrow[ci * 3 + 2] * pp;
        }
        acc += fmaxf(s, 0.f);
    }
    part[tid] = acc;
    __syncthreads();
    if ((tid & 1) == 0) {
        float m = (part[tid] + part[tid + 1]) * (1.f / 64.f);
        featbf[(size_t)node * 64 + cout] = f2bf(m);
    }
}

// ---- WMMA GEMM: C(M,N) = A(M,K)bf16 * Bt(N,K)bf16 (+bias,relu) ----
// Each wave computes a 16x64 strip (4 accumulators) so the A fragment is
// loaded once per k-step and reused by 4 v_wmma ops. N must be a multiple
// of 64, K a multiple of 32, M a multiple of 16.
__global__ void __launch_bounds__(256)
k_wmma_gemm(const bf16_t* __restrict__ A, const bf16_t* __restrict__ Bt,
            const float* __restrict__ bias, float* __restrict__ Cmat,
            int M, int N, int K, int fuse_bias_relu) {
    const int wave = (blockIdx.x * blockDim.x + threadIdx.x) >> 5;
    const int lane = threadIdx.x & 31;
    const int nstrips = N >> 6;                 // strips of 64 columns
    const int strip = wave % nstrips;
    const int tileM = wave / nstrips;
    if (tileM >= (M >> 4)) return;              // uniform per wave -> EXEC stays all-ones

    const int m = (tileM << 4) + (lane & 15);
    const int ncol = (lane & 15);
    const int offA = (lane < 16) ? 0 : 16;      // contiguous K order, matches B
    const int offB = (lane < 16) ? 0 : 16;      // contiguous K order, matches A
    const bf16_t* arow = A + (size_t)m * K;
    const bf16_t* brow0 = Bt + (size_t)((strip << 6) +  0 + ncol) * K + offB;
    const bf16_t* brow1 = Bt + (size_t)((strip << 6) + 16 + ncol) * K + offB;
    const bf16_t* brow2 = Bt + (size_t)((strip << 6) + 32 + ncol) * K + offB;
    const bf16_t* brow3 = Bt + (size_t)((strip << 6) + 48 + ncol) * K + offB;

    v8f acc0 = {}, acc1 = {}, acc2 = {}, acc3 = {};
    for (int k0 = 0; k0 < K; k0 += 32) {
        v8bf alo = *(const v8bf*)(arow + k0 + offA);        // K = off..off+7
        v8bf ahi = *(const v8bf*)(arow + k0 + offA + 8);    // K = off+8..off+15
        v16bf afrag = __builtin_shufflevector(alo, ahi,
                        0,1,2,3,4,5,6,7,8,9,10,11,12,13,14,15);
        v16bf b0 = *(const v16bf*)(brow0 + k0);             // 16 consecutive K
        v16bf b1 = *(const v16bf*)(brow1 + k0);
        v16bf b2 = *(const v16bf*)(brow2 + k0);
        v16bf b3 = *(const v16bf*)(brow3 + k0);
        acc0 = __builtin_amdgcn_wmma_f32_16x16x32_bf16(false, afrag, false, b0, (short)0, acc0, false, false);
        acc1 = __builtin_amdgcn_wmma_f32_16x16x32_bf16(false, afrag, false, b1, (short)0, acc1, false, false);
        acc2 = __builtin_amdgcn_wmma_f32_16x16x32_bf16(false, afrag, false, b2, (short)0, acc2, false, false);
        acc3 = __builtin_amdgcn_wmma_f32_16x16x32_bf16(false, afrag, false, b3, (short)0, acc3, false, false);
    }

    const int rbase = (tileM << 4) + ((lane >> 4) << 3);
    v8f accs[4] = {acc0, acc1, acc2, acc3};
#pragma unroll
    for (int t = 0; t < 4; ++t) {
        const int col = (strip << 6) + (t << 4) + ncol;
        const float bv = fuse_bias_relu ? bias[col] : 0.f;
#pragma unroll
        for (int r = 0; r < 8; ++r) {
            float v = accs[t][r];
            if (fuse_bias_relu) v = fmaxf(v + bv, 0.f);
            Cmat[(size_t)(rbase + r) * N + col] = v;
        }
    }
}

// ---------------- attention scalars: a_s,a_d per (b,n,h) ----------------
__global__ void k_att(const float* __restrict__ XL,
                      const float* __restrict__ as_, const float* __restrict__ ad_,
                      float* __restrict__ AS, float* __restrict__ AD) {
    int idx = blockIdx.x * blockDim.x + threadIdx.x;
    if (idx >= BN * HH) return;
    int h = idx & 3, bn = idx >> 2;
    const float* row = XL + (size_t)bn * GD + h * CC;
    float ss = 0.f, sd = 0.f;
    for (int c = 0; c < CC; ++c) {
        float v = row[c];
        ss += v * as_[h * CC + c];
        sd += v * ad_[h * CC + c];
    }
    AS[idx] = ss;
    AD[idx] = sd;
}

// ---------------- edge pass 1: segment max (ordered-uint atomicMax) ----------------
__global__ void k_edge_max(const int* __restrict__ ei,
                           const float* __restrict__ AS, const float* __restrict__ AD,
                           unsigned* __restrict__ EMAX) {
    int idx = blockIdx.x * blockDim.x + threadIdx.x;
    if (idx >= BB * EALL) return;
    int b = idx / EALL, e = idx % EALL;
    int s, d;
    if (e < EE) { s = ei[e]; d = ei[EE + e]; } else { s = d = e - EE; }
    int sb = (b * NN + s) * HH, db = (b * NN + d) * HH;
    for (int h = 0; h < HH; ++h) {
        float ev = lrelu(AS[sb + h] + AD[db + h]);
        atomicMax(&EMAX[db + h], f2ord(ev));
    }
}

// ---------------- edge pass 2: segment sum of exp ----------------
__global__ void k_edge_sum(const int* __restrict__ ei,
                           const float* __restrict__ AS, const float* __restrict__ AD,
                           const unsigned* __restrict__ EMAX, float* __restrict__ DEN) {
    int idx = blockIdx.x * blockDim.x + threadIdx.x;
    if (idx >= BB * EALL) return;
    int b = idx / EALL, e = idx % EALL;
    int s, d;
    if (e < EE) { s = ei[e]; d = ei[EE + e]; } else { s = d = e - EE; }
    int sb = (b * NN + s) * HH, db = (b * NN + d) * HH;
    for (int h = 0; h < HH; ++h) {
        float ev = lrelu(AS[sb + h] + AD[db + h]);
        float ex = __expf(ev - ord2f(EMAX[db + h]));
        atomAddF(&DEN[db + h], ex);
    }
}

// ---------------- edge pass 3: message scatter (256 threads per edge) ----------------
__global__ void k_edge_scatter(const int* __restrict__ ei,
                               const float* __restrict__ AS, const float* __restrict__ AD,
                               const unsigned* __restrict__ EMAX, const float* __restrict__ DEN,
                               const float* __restrict__ XL, float* __restrict__ AGG) {
    long long gid = (long long)blockIdx.x * blockDim.x + threadIdx.x;
    if (gid >= (long long)BB * EALL * GD) return;
    int c  = (int)(gid & 63);
    int h  = (int)((gid >> 6) & 3);
    long long be = gid >> 8;
    int b = (int)(be / EALL), e = (int)(be % EALL);
    int s, d;
    if (e < EE) { s = ei[e]; d = ei[EE + e]; } else { s = d = e - EE; }
    int sb = (b * NN + s) * HH, db = (b * NN + d) * HH;
    float ev    = lrelu(AS[sb + h] + AD[db + h]);
    float alpha = __expf(ev - ord2f(EMAX[db + h])) / (DEN[db + h] + 1e-16f);
    float val   = XL[(size_t)(b * NN + s) * GD + h * CC + c] * alpha;
    atomAddF(&AGG[(size_t)(b * NN + d) * GD + h * CC + c], val);
}

// ---------------- relu(agg + bias) -> f32 in place + bf16 copy ----------------
__global__ void k_relu_bias(float* __restrict__ AGG, const float* __restrict__ bias,
                            unsigned short* __restrict__ GBF) {
    size_t idx = (size_t)blockIdx.x * blockDim.x + threadIdx.x;
    if (idx >= (size_t)BN * GD) return;
    float v = fmaxf(AGG[idx] + bias[idx & (GD - 1)], 0.f);
    AGG[idx] = v;
    GBF[idx] = f2bf(v);
}

// ---------------- mean over nodes: (B,N,256) -> (B,256) ----------------
__global__ void __launch_bounds__(256)
k_pool_mean(const float* __restrict__ G, float* __restrict__ pooled) {
    int b = blockIdx.x, k = threadIdx.x;
    float s = 0.f;
    const float* base = G + (size_t)b * NN * GD + k;
    for (int n = 0; n < NN; ++n) s += base[(size_t)n * GD];
    pooled[b * GD + k] = s * (1.f / NN);
}

// ---------------- classification head: 256->128->2 ----------------
__global__ void __launch_bounds__(128)
k_class_head(const float* __restrict__ pooled,
             const float* __restrict__ Wc1, const float* __restrict__ bc1,
             const float* __restrict__ Wc2, const float* __restrict__ bc2,
             float* __restrict__ out) {
    __shared__ float hc[128];
    int j = threadIdx.x;
    for (int b = 0; b < BB; ++b) {
        float s = bc1[j];
        for (int k = 0; k < GD; ++k) s += pooled[b * GD + k] * Wc1[k * 128 + j];
        hc[j] = fmaxf(s, 0.f);
        __syncthreads();
        if (j < 2) {
            float s2 = bc2[j];
            for (int q = 0; q < 128; ++q) s2 += hc[q] * Wc2[q * 2 + j];
            out[b * 2 + j] = s2;
        }
        __syncthreads();
    }
}

// ---------------- attribution head final: sigmoid(H(BN,64)@Wa2 + ba2) ----------------
__global__ void k_attr_final(const float* __restrict__ Hm, const float* __restrict__ Wa2,
                             const float* __restrict__ ba2, float* __restrict__ out) {
    int idx = blockIdx.x * blockDim.x + threadIdx.x;
    if (idx >= BN) return;
    float s = ba2[0];
    const float* row = Hm + (size_t)idx * 64;
    for (int jj = 0; jj < 64; ++jj) s += row[jj] * Wa2[jj];
    out[idx] = 1.f / (1.f + __expf(-s));
}

// ---------------- host launch ----------------
extern "C" void kernel_launch(void* const* d_in, const int* in_sizes, int n_in,
                              void* d_out, int out_size, void* d_ws, size_t ws_size,
                              hipStream_t stream) {
    const float* x   = (const float*)d_in[0];
    const int*   ei  = (const int*)d_in[1];
    const float* cw1 = (const float*)d_in[2];
    const float* cb1 = (const float*)d_in[3];
    const float* cw2 = (const float*)d_in[4];
    const float* cb2 = (const float*)d_in[5];
    const float* W0  = (const float*)d_in[6];
    const float* as0 = (const float*)d_in[7];
    const float* ad0 = (const float*)d_in[8];
    const float* b0  = (const float*)d_in[9];
    const float* W1  = (const float*)d_in[10];
    const float* as1 = (const float*)d_in[11];
    const float* ad1 = (const float*)d_in[12];
    const float* b1  = (const float*)d_in[13];
    const float* Wc1 = (const float*)d_in[14];
    const float* bc1 = (const float*)d_in[15];
    const float* Wc2 = (const float*)d_in[16];
    const float* bc2 = (const float*)d_in[17];
    const float* Wa1 = (const float*)d_in[18];
    const float* ba1 = (const float*)d_in[19];
    const float* Wa2 = (const float*)d_in[20];
    const float* ba2 = (const float*)d_in[21];
    float* out = (float*)d_out;

    char* ws = (char*)d_ws;
    size_t off = 0;
    auto alloc = [&](size_t bytes) { size_t o = off; off += (bytes + 255) & ~(size_t)255; return o; };

    const size_t oXL    = alloc((size_t)BN * GD * 4);   // projections / HATTR reuse
    const size_t oAGG   = alloc((size_t)BN * GD * 4);   // aggregation / g f32
    const size_t oGBF   = alloc((size_t)BN * GD * 2);   // bf16 activations
    const size_t oFEAT  = alloc((size_t)BN * 64 * 2);   // bf16 conv features
    const size_t oW0T   = alloc((size_t)GD * 64 * 2);
    const size_t oW1T   = alloc((size_t)GD * GD * 2);
    const size_t oWA1T  = alloc((size_t)64 * GD * 2);
    const size_t oAS    = alloc((size_t)BN * HH * 4);
    const size_t oAD    = alloc((size_t)BN * HH * 4);
    const size_t oEMAX  = alloc((size_t)BN * HH * 4);
    const size_t oDEN   = alloc((size_t)BN * HH * 4);
    const size_t oPOOL  = alloc((size_t)BB * GD * 4);
    (void)ws_size; (void)in_sizes; (void)n_in; (void)out_size;

    float*          XL   = (float*)(ws + oXL);
    float*          AGG  = (float*)(ws + oAGG);
    unsigned short* GBF  = (unsigned short*)(ws + oGBF);
    unsigned short* FEAT = (unsigned short*)(ws + oFEAT);
    unsigned short* W0T  = (unsigned short*)(ws + oW0T);
    unsigned short* W1T  = (unsigned short*)(ws + oW1T);
    unsigned short* WA1T = (unsigned short*)(ws + oWA1T);
    float*          AS   = (float*)(ws + oAS);
    float*          AD   = (float*)(ws + oAD);
    unsigned*       EMAX = (unsigned*)(ws + oEMAX);
    float*          DEN  = (float*)(ws + oDEN);
    float*          POOL = (float*)(ws + oPOOL);

    // ---- weight transposes (fp32 -> bf16 (N,K)) ----
    k_transpose_bf16<<<(64 * GD + 255) / 256, 256, 0, stream>>>(W0, W0T, 64, GD);
    k_transpose_bf16<<<(GD * GD + 255) / 256, 256, 0, stream>>>(W1, W1T, GD, GD);
    k_transpose_bf16<<<(GD * 64 + 255) / 256, 256, 0, stream>>>(Wa1, WA1T, GD, 64);

    // ---- temporal conv -> bf16 features ----
    k_conv_temporal<<<BN, 128, 0, stream>>>(x, cw1, cb1, cw2, cb2, FEAT);

    const int edgeBlocks    = (BB * EALL + 255) / 256;
    const int scatterBlocks = (int)(((long long)BB * EALL * GD + 255) / 256);
    const int ewBlocks      = (BN * GD + 255) / 256;
    const int gemmBlocks256 = ((BN / 16) * (GD / 64) + 7) / 8;   // 16x64 strips, 8 waves/block
    const int gemmBlocks64  = ((BN / 16) * (64 / 64) + 7) / 8;

    // ================= GAT layer 0 =================
    k_wmma_gemm<<<gemmBlocks256, 256, 0, stream>>>(
        (const bf16_t*)FEAT, (const bf16_t*)W0T, nullptr, XL, BN, GD, 64, 0);
    hipMemsetAsync(AGG, 0, (size_t)BN * GD * 4, stream);
    hipMemsetAsync(EMAX, 0, (size_t)BN * HH * 4, stream);
    hipMemsetAsync(DEN, 0, (size_t)BN * HH * 4, stream);
    k_att<<<(BN * HH + 255) / 256, 256, 0, stream>>>(XL, as0, ad0, AS, AD);
    k_edge_max<<<edgeBlocks, 256, 0, stream>>>(ei, AS, AD, EMAX);
    k_edge_sum<<<edgeBlocks, 256, 0, stream>>>(ei, AS, AD, EMAX, DEN);
    k_edge_scatter<<<scatterBlocks, 256, 0, stream>>>(ei, AS, AD, EMAX, DEN, XL, AGG);
    k_relu_bias<<<ewBlocks, 256, 0, stream>>>(AGG, b0, GBF);

    // ================= GAT layer 1 =================
    k_wmma_gemm<<<gemmBlocks256, 256, 0, stream>>>(
        (const bf16_t*)GBF, (const bf16_t*)W1T, nullptr, XL, BN, GD, GD, 0);
    hipMemsetAsync(AGG, 0, (size_t)BN * GD * 4, stream);
    hipMemsetAsync(EMAX, 0, (size_t)BN * HH * 4, stream);
    hipMemsetAsync(DEN, 0, (size_t)BN * HH * 4, stream);
    k_att<<<(BN * HH + 255) / 256, 256, 0, stream>>>(XL, as1, ad1, AS, AD);
    k_edge_max<<<edgeBlocks, 256, 0, stream>>>(ei, AS, AD, EMAX);
    k_edge_sum<<<edgeBlocks, 256, 0, stream>>>(ei, AS, AD, EMAX, DEN);
    k_edge_scatter<<<scatterBlocks, 256, 0, stream>>>(ei, AS, AD, EMAX, DEN, XL, AGG);
    k_relu_bias<<<ewBlocks, 256, 0, stream>>>(AGG, b1, GBF);

    // ================= heads =================
    k_pool_mean<<<BB, GD, 0, stream>>>(AGG, POOL);
    k_class_head<<<1, 128, 0, stream>>>(POOL, Wc1, bc1, Wc2, bc2, out);

    // attribution: H = relu(g @ Wa1 + ba1) via WMMA (bias+relu fused), reuse XL
    k_wmma_gemm<<<gemmBlocks64, 256, 0, stream>>>(
        (const bf16_t*)GBF, (const bf16_t*)WA1T, ba1, XL, BN, 64, GD, 1);
    k_attr_final<<<(BN + 255) / 256, 256, 0, stream>>>(XL, Wa2, ba2, out + 8);
}
